// BertSelfAttention_76201309766513
// MI455X (gfx1250) — compile-verified, matrix operand order checked by reading, not run on hardware
//
#include <hip/hip_runtime.h>
#include <stdint.h>

#define HIDDEN 1024
#define NHEADS 16
#define HDIM   64
#define BATCH  8
#define SEQ    1024
#define MTOT   (BATCH*SEQ)   // 8192

typedef __attribute__((ext_vector_type(16))) __bf16 v16bf;
typedef __attribute__((ext_vector_type(8)))  __bf16 v8bf;
typedef __attribute__((ext_vector_type(8)))  float  v8f;

// round-to-nearest-even fp32 -> bf16 (pure integer)
__device__ __forceinline__ __bf16 f2bf(float x) {
  union { float f; uint32_t u; } v; v.f = x;
  uint32_t r = v.u + 0x7FFFu + ((v.u >> 16) & 1u);
  union { uint16_t s; __bf16 b; } o; o.s = (uint16_t)(r >> 16);
  return o.b;
}

// 16x32 bf16 fragment in the CDNA5 A/B lane layout:
//   lanes 0-15 : row = lane,     K = {0..7, 16..23}
//   lanes 16-31: row = lane-16,  K = {8..15, 24..31}
__device__ __forceinline__ v16bf load_frag(const __bf16* __restrict__ base,
                                           int stride, int lane) {
  const int r    = lane & 15;
  const int koff = (lane >> 4) << 3;
  const __bf16* p = base + (size_t)r * stride + koff;
  v8bf lo = *(const v8bf*)(p);
  v8bf hi = *(const v8bf*)(p + 16);
  v16bf out;
#pragma unroll
  for (int i = 0; i < 8; ++i) { out[i] = lo[i]; out[i + 8] = hi[i]; }
  return out;
}

__device__ __forceinline__ v8f wmma_bf16(v16bf a, v16bf b, v8f c) {
  return __builtin_amdgcn_wmma_f32_16x16x32_bf16(false, a, false, b, (short)0, c,
                                                 false, false);
}

__global__ void __launch_bounds__(256)
cvt_kernel(const float* __restrict__ src, __bf16* __restrict__ dst, int n) {
  int i = (blockIdx.x * blockDim.x + threadIdx.x) * 4;
  if (i + 3 < n) {
    float4 f = *(const float4*)(src + i);
    dst[i]     = f2bf(f.x);
    dst[i + 1] = f2bf(f.y);
    dst[i + 2] = f2bf(f.z);
    dst[i + 3] = f2bf(f.w);
  }
}

// grid: (HIDDEN/64, MTOT/128, 3), block 128 (4 waves).
// Wave owns a 32x64 strip (2 M-tiles x 4 N-tiles = 8 accumulators).
// Ping-pong double buffering, unrolled x2 so no register rotation copies:
// loads for one buffer stay in flight under the WMMAs of the other.
__global__ void __launch_bounds__(128)
qkv_gemm_kernel(const __bf16* __restrict__ X, const __bf16* __restrict__ W3,
                const float* __restrict__ bq, const float* __restrict__ bk,
                const float* __restrict__ bvp,
                __bf16* __restrict__ Q, __bf16* __restrict__ K,
                __bf16* __restrict__ Vt) {
  const int lane = threadIdx.x & 31;
  const int wave = threadIdx.x >> 5;
  const int n0   = blockIdx.x * 64;
  const int m0   = blockIdx.y * 128 + wave * 32;
  const int mat  = blockIdx.z;

  const __bf16* W   = W3 + (size_t)mat * HIDDEN * HIDDEN;
  const float* bias = (mat == 0) ? bq : (mat == 1) ? bk : bvp;

  const __bf16* ar0 = X + (size_t)m0 * HIDDEN;
  const __bf16* ar1 = ar0 + (size_t)16 * HIDDEN;
  const __bf16* wr0 = W + (size_t)(n0 +  0) * HIDDEN;
  const __bf16* wr1 = W + (size_t)(n0 + 16) * HIDDEN;
  const __bf16* wr2 = W + (size_t)(n0 + 32) * HIDDEN;
  const __bf16* wr3 = W + (size_t)(n0 + 48) * HIDDEN;

  v8f acc[2][4];
#pragma unroll
  for (int mt = 0; mt < 2; ++mt)
#pragma unroll
    for (int t = 0; t < 4; ++t) acc[mt][t] = (v8f){};

  // buffer set 0: k = 0
  v16bf a00 = load_frag(ar0, HIDDEN, lane);
  v16bf a01 = load_frag(ar1, HIDDEN, lane);
  v16bf b00 = load_frag(wr0, HIDDEN, lane);
  v16bf b01 = load_frag(wr1, HIDDEN, lane);
  v16bf b02 = load_frag(wr2, HIDDEN, lane);
  v16bf b03 = load_frag(wr3, HIDDEN, lane);

#pragma unroll 1
  for (int k0 = 0; k0 < HIDDEN; k0 += 64) {
    // --- load buffer set 1 @ k0+32 (always in range: k0 <= 960) ---
    const int k1 = k0 + 32;
    v16bf a10 = load_frag(ar0 + k1, HIDDEN, lane);
    v16bf a11 = load_frag(ar1 + k1, HIDDEN, lane);
    v16bf b10 = load_frag(wr0 + k1, HIDDEN, lane);
    v16bf b11 = load_frag(wr1 + k1, HIDDEN, lane);
    v16bf b12 = load_frag(wr2 + k1, HIDDEN, lane);
    v16bf b13 = load_frag(wr3 + k1, HIDDEN, lane);

    // --- WMMAs on buffer set 0 (overlap set-1 loads) ---
    acc[0][0] = wmma_bf16(a00, b00, acc[0][0]);
    acc[1][0] = wmma_bf16(a01, b00, acc[1][0]);
    acc[0][1] = wmma_bf16(a00, b01, acc[0][1]);
    acc[1][1] = wmma_bf16(a01, b01, acc[1][1]);
    acc[0][2] = wmma_bf16(a00, b02, acc[0][2]);
    acc[1][2] = wmma_bf16(a01, b02, acc[1][2]);
    acc[0][3] = wmma_bf16(a00, b03, acc[0][3]);
    acc[1][3] = wmma_bf16(a01, b03, acc[1][3]);

    // --- load buffer set 0 @ k0+64 (wraps to dummy 0 on last iter) ---
    const int k2 = (k0 + 64) & (HIDDEN - 1);
    a00 = load_frag(ar0 + k2, HIDDEN, lane);
    a01 = load_frag(ar1 + k2, HIDDEN, lane);
    b00 = load_frag(wr0 + k2, HIDDEN, lane);
    b01 = load_frag(wr1 + k2, HIDDEN, lane);
    b02 = load_frag(wr2 + k2, HIDDEN, lane);
    b03 = load_frag(wr3 + k2, HIDDEN, lane);

    // --- WMMAs on buffer set 1 (overlap set-0 loads) ---
    acc[0][0] = wmma_bf16(a10, b10, acc[0][0]);
    acc[1][0] = wmma_bf16(a11, b10, acc[1][0]);
    acc[0][1] = wmma_bf16(a10, b11, acc[0][1]);
    acc[1][1] = wmma_bf16(a11, b11, acc[1][1]);
    acc[0][2] = wmma_bf16(a10, b12, acc[0][2]);
    acc[1][2] = wmma_bf16(a11, b12, acc[1][2]);
    acc[0][3] = wmma_bf16(a10, b13, acc[0][3]);
    acc[1][3] = wmma_bf16(a11, b13, acc[1][3]);
  }

  const int hiL = lane >> 4, lo = lane & 15;
#pragma unroll
  for (int mt = 0; mt < 2; ++mt) {
#pragma unroll
    for (int t = 0; t < 4; ++t) {
      const int col = n0 + t * 16 + lo;
      const int h = col >> 6, d = col & 63;
      const float bb = bias[col];
#pragma unroll
      for (int j = 0; j < 8; ++j) {
        const int m = m0 + mt * 16 + j + 8 * hiL;   // global row (b*SEQ + s)
        const int bat = m >> 10, s = m & 1023;
        float val = acc[mt][t][j] + bb;
        const size_t bh = (size_t)(bat * NHEADS + h);
        if (mat == 0) {
          val *= 0.125f;                            // 1/sqrt(HDIM) folded into Q
          Q[(bh * SEQ + s) * HDIM + d] = f2bf(val);
        } else if (mat == 1) {
          K[(bh * SEQ + s) * HDIM + d] = f2bf(val);
        } else {
          Vt[(bh * HDIM + d) * SEQ + s] = f2bf(val); // V stored transposed
        }
      }
    }
  }
}

// grid: (SEQ/64, BATCH*NHEADS), block 128 (4 waves). Wave owns 16 query rows.
// Flash attention, online softmax. Two 32-wide key blocks per loop iteration
// with ping-pong K-fragment buffers (no rotation copies); V fragments hoisted
// above the softmax so exp/shuffle VALU work hides their latency.
__global__ void __launch_bounds__(128)
attn_kernel(const __bf16* __restrict__ Q, const __bf16* __restrict__ K,
            const __bf16* __restrict__ Vt, const float* __restrict__ mask,
            float* __restrict__ out) {
  __shared__ __align__(16) __bf16 plds[4][16][40];   // 40*2B row stride: 16B aligned
  const int lane = threadIdx.x & 31;
  const int wave = threadIdx.x >> 5;
  const int bh = blockIdx.y;
  const int bat = bh >> 4, h = bh & 15;
  const int q0 = blockIdx.x * 64 + wave * 16;
  const int hiL = lane >> 4, lo = lane & 15;

  const __bf16* Qb = Q + ((size_t)bh * SEQ + q0) * HDIM;
  const v16bf qa0 = load_frag(Qb, HDIM, lane);        // d = 0..31
  const v16bf qa1 = load_frag(Qb + 32, HDIM, lane);   // d = 32..63

  v8f o0 = {}, o1 = {}, o2 = {}, o3 = {};
  float mst[8], lst[8];
#pragma unroll
  for (int j = 0; j < 8; ++j) { mst[j] = -3.0e38f; lst[j] = 0.f; }

  const float* mrow = mask + (size_t)bat * SEQ;
  __bf16* pw = &plds[wave][0][0];
  const __bf16* Kbh = K + (size_t)bh * SEQ * HDIM;
  const __bf16* Vbh = Vt + (size_t)bh * HDIM * SEQ;

  v16bf ka[4], kb[4];
  // prologue: K fragments for kk = 0 into set A
  ka[0] = load_frag(Kbh, HDIM, lane);
  ka[1] = load_frag(Kbh + 32, HDIM, lane);
  ka[2] = load_frag(Kbh + 16 * HDIM, HDIM, lane);
  ka[3] = load_frag(Kbh + 16 * HDIM + 32, HDIM, lane);

  // One 32-wide key block: consumes cur[], preloads nxt[] for kkc+32.
  auto phase = [&](int kkc, v16bf* cur, v16bf* nxt) {
    // V fragments for this block: independent of softmax, issue early
    const __bf16* Vb = Vbh + kkc;
    v16bf vf0 = load_frag(Vb,            SEQ, lane);
    v16bf vf1 = load_frag(Vb + 16 * SEQ, SEQ, lane);
    v16bf vf2 = load_frag(Vb + 32 * SEQ, SEQ, lane);
    v16bf vf3 = load_frag(Vb + 48 * SEQ, SEQ, lane);

    // scores: two 16x16 N-tiles, each reduced over d=64
    v8f s0 = {}, s1 = {};
    s0 = wmma_bf16(qa0, cur[0], s0);
    s0 = wmma_bf16(qa1, cur[1], s0);
    s1 = wmma_bf16(qa0, cur[2], s1);
    s1 = wmma_bf16(qa1, cur[3], s1);

    // next block's K fragments (dummy wrapped load on final phase)
    const int kn = (kkc + 32) & (SEQ - 1);
    const __bf16* Kn = Kbh + (size_t)kn * HDIM;
    nxt[0] = load_frag(Kn, HDIM, lane);
    nxt[1] = load_frag(Kn + 32, HDIM, lane);
    nxt[2] = load_frag(Kn + 16 * HDIM, HDIM, lane);
    nxt[3] = load_frag(Kn + 16 * HDIM + 32, HDIM, lane);

    const float mk0 = mrow[kkc + lo];
    const float mk1 = mrow[kkc + 16 + lo];

    // D layout: element j = row (j + 8*(lane>>4)), col (lane&15).
    // Row reductions via xor 1/2/4/8 stay within each 16-lane half.
#pragma unroll
    for (int j = 0; j < 8; ++j) {
      float a = s0[j] + mk0;
      float c = s1[j] + mk1;
      float mx = fmaxf(a, c);
      mx = fmaxf(mx, __shfl_xor(mx, 1));
      mx = fmaxf(mx, __shfl_xor(mx, 2));
      mx = fmaxf(mx, __shfl_xor(mx, 4));
      mx = fmaxf(mx, __shfl_xor(mx, 8));
      const float mnew = fmaxf(mst[j], mx);
      const float corr = __expf(mst[j] - mnew);
      const float p0 = __expf(a - mnew);
      const float p1 = __expf(c - mnew);
      float rs = p0 + p1;
      rs += __shfl_xor(rs, 1);
      rs += __shfl_xor(rs, 2);
      rs += __shfl_xor(rs, 4);
      rs += __shfl_xor(rs, 8);
      lst[j] = lst[j] * corr + rs;
      mst[j] = mnew;
      o0[j] *= corr; o1[j] *= corr; o2[j] *= corr; o3[j] *= corr;
      // stage P (D layout) into LDS for transpose to A layout
      pw[(j + 8 * hiL) * 40 + lo]      = f2bf(p0);
      pw[(j + 8 * hiL) * 40 + 16 + lo] = f2bf(p1);
    }
    __builtin_amdgcn_wave_barrier();   // DS ops are wave-in-order; block reordering only
    const v16bf pf = load_frag(pw, 40, lane);   // P as 16x32 A fragment
    __builtin_amdgcn_wave_barrier();

    // ctx += P(16x32) x V(32x64)
    o0 = wmma_bf16(pf, vf0, o0);
    o1 = wmma_bf16(pf, vf1, o1);
    o2 = wmma_bf16(pf, vf2, o2);
    o3 = wmma_bf16(pf, vf3, o3);
  };

#pragma unroll 1
  for (int kk = 0; kk < SEQ; kk += 64) {
    phase(kk,      ka, kb);   // consumes A, fills B
    phase(kk + 32, kb, ka);   // consumes B, fills A
  }

#pragma unroll
  for (int j = 0; j < 8; ++j) {
    const int qrow = q0 + j + 8 * hiL;
    const float inv = 1.0f / lst[j];
    float* orow = out + ((size_t)bat * SEQ + qrow) * HIDDEN + h * HDIM;
    orow[lo]      = o0[j] * inv;
    orow[16 + lo] = o1[j] * inv;
    orow[32 + lo] = o2[j] * inv;
    orow[48 + lo] = o3[j] * inv;
  }
}

extern "C" void kernel_launch(void* const* d_in, const int* in_sizes, int n_in,
                              void* d_out, int out_size, void* d_ws, size_t ws_size,
                              hipStream_t stream) {
  (void)in_sizes; (void)n_in; (void)out_size; (void)ws_size;
  const float* X    = (const float*)d_in[0];
  const float* mask = (const float*)d_in[1];
  const float* Wq   = (const float*)d_in[2];
  const float* bq   = (const float*)d_in[3];
  const float* Wk   = (const float*)d_in[4];
  const float* bk   = (const float*)d_in[5];
  const float* Wv   = (const float*)d_in[6];
  const float* bv   = (const float*)d_in[7];

  // Workspace layout (bf16): X 16MiB | W 6MiB | Q 16MiB | K 16MiB | Vt 16MiB = 70MiB
  __bf16* Xb  = (__bf16*)d_ws;
  __bf16* Wb  = Xb + (size_t)MTOT * HIDDEN;
  __bf16* Qb  = Wb + (size_t)3 * HIDDEN * HIDDEN;
  __bf16* Kb  = Qb + (size_t)BATCH * NHEADS * SEQ * HDIM;
  __bf16* Vtb = Kb + (size_t)BATCH * NHEADS * SEQ * HDIM;

  const int nx = MTOT * HIDDEN;       // 8388608
  const int nw = HIDDEN * HIDDEN;     // 1048576
  cvt_kernel<<<nx / 1024, 256, 0, stream>>>(X,  Xb, nx);
  cvt_kernel<<<nw / 1024, 256, 0, stream>>>(Wq, Wb, nw);
  cvt_kernel<<<nw / 1024, 256, 0, stream>>>(Wk, Wb + nw, nw);
  cvt_kernel<<<nw / 1024, 256, 0, stream>>>(Wv, Wb + 2 * nw, nw);

  qkv_gemm_kernel<<<dim3(HIDDEN / 64, MTOT / 128, 3), 128, 0, stream>>>(
      Xb, Wb, bq, bk, bv, Qb, Kb, Vtb);

  attn_kernel<<<dim3(SEQ / 64, BATCH * NHEADS), 128, 0, stream>>>(
      Qb, Kb, Vtb, mask, (float*)d_out);
}
